// MSAEWrapper_27788438405443
// MI455X (gfx1250) — matrix operand order repaired and checked by reading, not compiled, not run on hardware
//
#include <hip/hip_runtime.h>
#include <hip/hip_bf16.h>

// MSAE forward: z = relu(((x-mc)*s - pb) @ enc + lb); sae = (z@dec + pb)/s + mc; topk(z,64)
// B=4096, D_IN=1024, D_SAE=16384, K=64.  bf16 WMMA (f32 accum), double-buffered LDS.

#define B_ROWS   4096
#define D_IN     1024
#define D_SAE    16384
#define TOPK     64

typedef __attribute__((ext_vector_type(16))) __bf16        v16bf;
typedef __attribute__((ext_vector_type(8)))  float         v8f;
typedef __attribute__((ext_vector_type(8)))  unsigned int  v8u;

// Packed f32x2 -> bf16x2 (round-to-nearest-even).  Prefer the HW packed convert.
__device__ __forceinline__ unsigned pk2(float lo, float hi) {
#if __has_builtin(__builtin_amdgcn_cvt_pk_bf16_f32)
  auto r = __builtin_amdgcn_cvt_pk_bf16_f32(lo, hi);
  return __builtin_bit_cast(unsigned, r);
#else
  const unsigned ul = __builtin_bit_cast(unsigned, lo);
  const unsigned uh = __builtin_bit_cast(unsigned, hi);
  const unsigned rl = (ul + 0x7FFFu + ((ul >> 16) & 1u)) >> 16;
  const unsigned rh = (uh + 0x7FFFu + ((uh >> 16) & 1u)) & 0xFFFF0000u;
  return rl | rh;
#endif
}

// A fragment (16x32 bf16, ISA layout): lane half 0 holds K=0..7 (v0..3) and 16..23 (v4..7),
// half 1 holds K=8..15 and 24..31.  A_lds is row-major [rows][32] bf16.
__device__ __forceinline__ v16bf ldsA_frag(const unsigned short* A_lds, int r, int half) {
  const unsigned short* p = A_lds + r * 32 + half * 8;
  const uint4 lo = *(const uint4*)p;          // K = half*8 + 0..7
  const uint4 hi = *(const uint4*)(p + 16);   // K = 16 + half*8 + 0..7
  v8u t;
  t[0] = lo.x; t[1] = lo.y; t[2] = lo.z; t[3] = lo.w;
  t[4] = hi.x; t[5] = hi.y; t[6] = hi.z; t[7] = hi.w;
  return __builtin_bit_cast(v16bf, t);
}
// B fragment (32x16 bf16): per lane (col n), half 0 holds K=0..7 and 16..23, half 1 K=8..15 and 24..31.
// B_ldsT is transposed [cols][32] bf16 so each frag is two runs of 8 contiguous bf16.
__device__ __forceinline__ v16bf ldsB_frag(const unsigned short* B_ldsT, int n, int half) {
  const unsigned short* p = B_ldsT + n * 32 + half * 8;
  const uint4 lo = *(const uint4*)p;
  const uint4 hi = *(const uint4*)(p + 16);
  v8u t;
  t[0] = lo.x; t[1] = lo.y; t[2] = lo.z; t[3] = lo.w;
  t[4] = hi.x; t[5] = hi.y; t[6] = hi.z; t[7] = hi.w;
  return __builtin_bit_cast(v16bf, t);
}

// ---------------- Fused GEMM (encode / decode) ----------------
// ENCODE:  out[z]  = relu(((A[x]-mc)*s - pb) @ Bm[enc] + lb)        KDIM=D_IN,  NDIM=D_SAE
// !ENCODE: out[sae]= (A[z] @ Bm[dec] + pb) / s + mc                 KDIM=D_SAE, NDIM=D_IN
// WG: 256 thr = 8 waves (4 M-waves x 2 N-waves), 128x128 tile, K step 32, LDS double-buffered.
template <int KDIM, int NDIM, bool ENCODE>
__global__ __launch_bounds__(256) void msae_gemm_wmma(
    const float* __restrict__ A, const float* __restrict__ Bm,
    const float* __restrict__ pre_bias, const float* __restrict__ lat_bias,
    const float* __restrict__ mean_center, const float* __restrict__ sfp,
    float* __restrict__ out)
{
  __shared__ unsigned short A_lds[2][128 * 32];   // 2 x 8 KB
  __shared__ unsigned short B_ldsT[2][128 * 32];  // 2 x 8 KB, transposed

  const int t     = threadIdx.x;
  const int lane  = t & 31;
  const int wave  = t >> 5;       // 0..7
  const int waveM = wave & 3;     // 4 waves along M (32 rows each)
  const int waveN = wave >> 2;    // 2 waves along N (64 cols each)
  const int m_blk = blockIdx.y * 128;
  const int n_blk = blockIdx.x * 128;
  const float s   = sfp[0];

  const int mload = t >> 3;        // 0..31 (A: 8 threads per row, float4 each)
  const int kload = (t & 7) * 4;   // A cols within chunk
  const int kpair = (t >> 5) * 2;  // 0,2,..,14 (B: paired K rows)
  const int nload = (t & 31) * 4;  // B cols, float4

  v8f acc[8] = {};

  const int nchunks = KDIM / 32;

  // ---- stage chunk 0 into buffer 0 ----
  {
    float4 mc4, pb4;
    if (ENCODE) { mc4 = *(const float4*)(mean_center + kload);
                  pb4 = *(const float4*)(pre_bias + kload); }
#pragma unroll
    for (int i = 0; i < 4; ++i) {
      const int r = mload + 32 * i;
      float4 v = *(const float4*)(A + (size_t)(m_blk + r) * KDIM + kload);
      if (ENCODE) {
        v.x = (v.x - mc4.x) * s - pb4.x;  v.y = (v.y - mc4.y) * s - pb4.y;
        v.z = (v.z - mc4.z) * s - pb4.z;  v.w = (v.w - mc4.w) * s - pb4.w;
      }
      *(uint2*)(&A_lds[0][r * 32 + kload]) = make_uint2(pk2(v.x, v.y), pk2(v.z, v.w));
    }
#pragma unroll
    for (int i = 0; i < 2; ++i) {
      const int kr = kpair + 16 * i;
      const float4 v0 = *(const float4*)(Bm + (size_t)(kr + 0) * NDIM + n_blk + nload);
      const float4 v1 = *(const float4*)(Bm + (size_t)(kr + 1) * NDIM + n_blk + nload);
      *(unsigned*)(&B_ldsT[0][(nload + 0) * 32 + kr]) = pk2(v0.x, v1.x);
      *(unsigned*)(&B_ldsT[0][(nload + 1) * 32 + kr]) = pk2(v0.y, v1.y);
      *(unsigned*)(&B_ldsT[0][(nload + 2) * 32 + kr]) = pk2(v0.z, v1.z);
      *(unsigned*)(&B_ldsT[0][(nload + 3) * 32 + kr]) = pk2(v0.w, v1.w);
    }
  }
  __syncthreads();

#pragma unroll 2
  for (int c = 0; c < nchunks; ++c) {
    const int cur = c & 1;
    const int nxt = cur ^ 1;

    // ---- stage chunk c+1 into the alternate buffer (overlaps with WMMA below) ----
    if (c + 1 < nchunks) {
      const int k0 = (c + 1) * 32;
      float4 mc4, pb4;
      if (ENCODE) { mc4 = *(const float4*)(mean_center + k0 + kload);
                    pb4 = *(const float4*)(pre_bias + k0 + kload); }
#pragma unroll
      for (int i = 0; i < 4; ++i) {
        const int r = mload + 32 * i;
        float4 v = *(const float4*)(A + (size_t)(m_blk + r) * KDIM + k0 + kload);
        if (ENCODE) {
          v.x = (v.x - mc4.x) * s - pb4.x;  v.y = (v.y - mc4.y) * s - pb4.y;
          v.z = (v.z - mc4.z) * s - pb4.z;  v.w = (v.w - mc4.w) * s - pb4.w;
        }
        *(uint2*)(&A_lds[nxt][r * 32 + kload]) = make_uint2(pk2(v.x, v.y), pk2(v.z, v.w));
      }
#pragma unroll
      for (int i = 0; i < 2; ++i) {
        const int kr = kpair + 16 * i;
        const float4 v0 = *(const float4*)(Bm + (size_t)(k0 + kr + 0) * NDIM + n_blk + nload);
        const float4 v1 = *(const float4*)(Bm + (size_t)(k0 + kr + 1) * NDIM + n_blk + nload);
        *(unsigned*)(&B_ldsT[nxt][(nload + 0) * 32 + kr]) = pk2(v0.x, v1.x);
        *(unsigned*)(&B_ldsT[nxt][(nload + 1) * 32 + kr]) = pk2(v0.y, v1.y);
        *(unsigned*)(&B_ldsT[nxt][(nload + 2) * 32 + kr]) = pk2(v0.z, v1.z);
        *(unsigned*)(&B_ldsT[nxt][(nload + 3) * 32 + kr]) = pk2(v0.w, v1.w);
      }
      if (c + 2 < nchunks) {  // L2 warm-up for chunk c+2 (global_prefetch_b8)
        __builtin_prefetch(A + (size_t)(m_blk + mload) * KDIM + (c + 2) * 32 + kload, 0, 0);
        __builtin_prefetch(Bm + (size_t)((c + 2) * 32 + kpair) * NDIM + n_blk + nload, 0, 0);
      }
    }

    // ---- compute on chunk c from the current buffer ----
    const int half = lane >> 4;
    const int ml   = lane & 15;
    v16bf afrag[2], bfrag[4];
#pragma unroll
    for (int tm = 0; tm < 2; ++tm)
      afrag[tm] = ldsA_frag(A_lds[cur], waveM * 32 + tm * 16 + ml, half);
#pragma unroll
    for (int tn = 0; tn < 4; ++tn)
      bfrag[tn] = ldsB_frag(B_ldsT[cur], waveN * 64 + tn * 16 + ml, half);
#pragma unroll
    for (int tm = 0; tm < 2; ++tm)
#pragma unroll
      for (int tn = 0; tn < 4; ++tn)
        acc[tm * 4 + tn] = __builtin_amdgcn_wmma_f32_16x16x32_bf16(
            false, afrag[tm], false, bfrag[tn], (short)0, acc[tm * 4 + tn], false, false);

    __syncthreads();
  }

  // ---- epilogue ----
  const int half = lane >> 4;
  const int nl   = lane & 15;
  const float inv_s = 1.0f / s;
#pragma unroll
  for (int tm = 0; tm < 2; ++tm)
#pragma unroll
    for (int tn = 0; tn < 4; ++tn) {
      const int col = n_blk + waveN * 64 + tn * 16 + nl;
      float b0, b1;
      if (ENCODE) { b0 = lat_bias[col]; }
      else        { b0 = pre_bias[col]; b1 = mean_center[col]; }
#pragma unroll
      for (int v = 0; v < 8; ++v) {
        const int row = m_blk + waveM * 32 + tm * 16 + v + 8 * half;
        const float a = acc[tm * 4 + tn][v];
        if (ENCODE) {
          const float r = a + b0;
          out[(size_t)row * NDIM + col] = r > 0.0f ? r : 0.0f;
        } else {
          out[(size_t)row * NDIM + col] = (a + b0) * inv_s + b1;
        }
      }
    }
}

// ---------------- Top-64 per row of z (positive-masked, JAX tie-break) ----------------
__global__ __launch_bounds__(256)
void msae_topk(const float* __restrict__ z, float* __restrict__ top_acts,
               int* __restrict__ top_idx)
{
  extern __shared__ char smem_raw[];
  float* row = (float*)smem_raw;                                         // 16384 f32
  unsigned long long* red = (unsigned long long*)(smem_raw + D_SAE * 4); // 256 u64

  const int b = blockIdx.x;
  const int t = threadIdx.x;
  const float* zr = z + (size_t)b * D_SAE;

  for (int i = 0; i < 64; ++i) row[i * 256 + t] = zr[i * 256 + t];
  __syncthreads();

  for (int sel = 0; sel < TOPK; ++sel) {
    // post-relu values are >= 0, so float-bit ordering == value ordering;
    // ~idx in the low word breaks ties toward the smaller index (JAX top_k order).
    unsigned long long best = 0;
#pragma unroll 8
    for (int i = 0; i < 64; ++i) {
      const int idx = i * 256 + t;
      const unsigned fb = __builtin_bit_cast(unsigned, row[idx]);
      const unsigned long long key =
          ((unsigned long long)fb << 32) | (unsigned)(~idx);
      best = key > best ? key : best;
    }
    red[t] = best;
    __syncthreads();
    for (int sred = 128; sred > 0; sred >>= 1) {
      if (t < sred) {
        const unsigned long long o = red[t + sred];
        if (o > red[t]) red[t] = o;
      }
      __syncthreads();
    }
    const unsigned long long win = red[0];
    const unsigned fb = (unsigned)(win >> 32);
    const int idx = (int)(~(unsigned)win);
    const float val = __builtin_bit_cast(float, fb);
    if (t == 0) {
      top_acts[(size_t)b * TOPK + sel] = val > 0.0f ? val : 0.0f;
      top_idx[(size_t)b * TOPK + sel]  = val > 0.0f ? idx : 0;
      row[idx] = 0.0f;  // mask winner (zero entries collapse to (0,0), same as reference)
    }
    __syncthreads();
  }
}

extern "C" void kernel_launch(void* const* d_in, const int* in_sizes, int n_in,
                              void* d_out, int out_size, void* d_ws, size_t ws_size,
                              hipStream_t stream) {
  (void)in_sizes; (void)n_in; (void)out_size; (void)d_ws; (void)ws_size;
  const float* x   = (const float*)d_in[0];
  const float* enc = (const float*)d_in[1];
  const float* dec = (const float*)d_in[2];
  const float* pb  = (const float*)d_in[3];
  const float* lb  = (const float*)d_in[4];
  const float* mc  = (const float*)d_in[5];
  const float* sf  = (const float*)d_in[6];
  // d_in[7] = k (fixed at 64)

  float* sae_out  = (float*)d_out;
  float* z        = sae_out + (size_t)B_ROWS * D_IN;
  float* top_acts = z + (size_t)B_ROWS * D_SAE;
  int*   top_idx  = (int*)(top_acts + (size_t)B_ROWS * TOPK);

  dim3 blk(256);
  msae_gemm_wmma<D_IN, D_SAE, true>
      <<<dim3(D_SAE / 128, B_ROWS / 128), blk, 0, stream>>>(
          x, enc, pb, lb, mc, sf, z);
  msae_gemm_wmma<D_SAE, D_IN, false>
      <<<dim3(D_IN / 128, B_ROWS / 128), blk, 0, stream>>>(
          z, dec, pb, lb, mc, sf, sae_out);
  msae_topk<<<dim3(B_ROWS), blk, D_SAE * 4 + 256 * 8, stream>>>(
      z, top_acts, top_idx);
}
